// NonLocalBlock_46548855554375
// MI455X (gfx1250) — matrix operand order repaired and checked by reading, not compiled, not run on hardware
//
#include <hip/hip_runtime.h>
#include <hip/hip_bf16.h>
#include <math.h>

typedef _Float16 h16;
typedef __attribute__((ext_vector_type(16))) _Float16 v16h;
typedef __attribute__((ext_vector_type(8)))  _Float16 v8h;
typedef __attribute__((ext_vector_type(8)))  float    v8f;

#define C_DIM 128
#define N_POS 4096
#define KT_STRIDE 136   // padded row stride (halves) for [j][c] tiles: bank-conflict-free
#define VC_STRIDE 40    // padded row stride (halves) for [c][j] tiles

// Contiguous-order fragment (unused): 16 contiguous halves (two 16B-aligned v8h loads)
__device__ __forceinline__ v16h ld16(const h16* p) {
    v8h lo = *(const v8h*)p;
    v8h hi = *(const v8h*)(p + 8);
    return __builtin_shufflevector(lo, hi, 0,1,2,3,4,5,6,7,8,9,10,11,12,13,14,15);
}
// A-layout fragment: halves at [p..p+7] and [p+16..p+23]
__device__ __forceinline__ v16h ld_a(const h16* p) {
    v8h lo = *(const v8h*)p;
    v8h hi = *(const v8h*)(p + 16);
    return __builtin_shufflevector(lo, hi, 0,1,2,3,4,5,6,7,8,9,10,11,12,13,14,15);
}
__device__ __forceinline__ v8f wmma(v16h a, v16h b, v8f c) {
    return __builtin_amdgcn_wmma_f32_16x16x32_f16(false, a, false, b, (short)0, c,
                                                  false, false);
}

// ---------------------------------------------------------------------------
// Kernel 0: convert the four weight matrices to f16 (one-time, 64K elements)
// ---------------------------------------------------------------------------
__global__ __launch_bounds__(256) void wconv_kernel(
    const float* __restrict__ Wq, const float* __restrict__ Wk,
    const float* __restrict__ Wv, const float* __restrict__ Wo,
    h16* __restrict__ Wq16, h16* __restrict__ Wk16,
    h16* __restrict__ Wv16, h16* __restrict__ Wo16)
{
    int idx = blockIdx.x * 256 + threadIdx.x;   // grid 64 -> 16384
    Wq16[idx] = (h16)Wq[idx];
    Wk16[idx] = (h16)Wk[idx];
    Wv16[idx] = (h16)Wv[idx];
    Wo16[idx] = (h16)Wo[idx];
}

// ---------------------------------------------------------------------------
// Kernel 1: QKV projections as WMMA GEMMs.
//   Qt[b][i][c], Kt[b][i][c] (position-major), Vc[b][c][i] (channel-major)
// Q/K computed with M=cout,N=pos (contiguous v8h stores into [i][c]);
// V computed with M=pos,N=cout (contiguous v8h stores into [c][i]).
// ---------------------------------------------------------------------------
__global__ __launch_bounds__(256) void qkv_proj_kernel(
    const float* __restrict__ x,
    const h16* __restrict__ Wq16, const float* __restrict__ bq,
    const h16* __restrict__ Wk16, const float* __restrict__ bk,
    const h16* __restrict__ Wv16, const float* __restrict__ bv,
    h16* __restrict__ Qt, h16* __restrict__ Kt, h16* __restrict__ Vc)
{
    const int b    = blockIdx.y;
    const int i0   = blockIdx.x * 64;
    const int lane = threadIdx.x & 31;
    const int wave = threadIdx.x >> 5;          // 8 waves
    const int m    = lane & 15;
    const int lh   = lane >> 4;

    __shared__ h16 xt[64 * KT_STRIDE];          // [i][cin] f16, padded

    for (int t = threadIdx.x; t < C_DIM * 64; t += 256) {
        int cin = t >> 6, ii = t & 63;          // coalesced global reads
        xt[ii * KT_STRIDE + cin] =
            (h16)x[((size_t)(b * C_DIM + cin)) * N_POS + i0 + ii];
    }
    __syncthreads();

    // 96 output tiles: 64 QK tiles (mat,ct,it) + 32 V tiles (it,ct)
    for (int tt = wave; tt < 96; tt += 8) {
        if (tt < 64) {
            int mat = tt >> 5, rest = tt & 31;
            int ct = rest >> 2, it = rest & 3;
            const h16*  W  = mat ? Wk16 : Wq16;
            const float* bs = mat ? bk : bq;
            v8f acc;
#pragma unroll
            for (int r = 0; r < 8; ++r) acc[r] = bs[ct * 16 + 8 * lh + r];
            const h16* wrow = W + (ct * 16 + m) * C_DIM + 8 * lh;
            const h16* xrow = xt + (it * 16 + m) * KT_STRIDE + 8 * lh;
#pragma unroll
            for (int c0 = 0; c0 < 4; ++c0)
                acc = wmma(ld_a(wrow + c0 * 32), ld_a(xrow + c0 * 32), acc);
            v8h st;
#pragma unroll
            for (int r = 0; r < 8; ++r) st[r] = (h16)acc[r];
            h16* dst = (mat ? Kt : Qt) +
                ((size_t)(b * N_POS + i0 + it * 16 + m)) * C_DIM + ct * 16 + 8 * lh;
            *(v8h*)dst = st;
        } else {
            int tt2 = tt - 64;
            int it = tt2 >> 3, ct = tt2 & 7;
            float bvv = bv[ct * 16 + m];
            v8f acc;
#pragma unroll
            for (int r = 0; r < 8; ++r) acc[r] = bvv;
            const h16* xrow = xt + (it * 16 + m) * KT_STRIDE + 8 * lh;
            const h16* wrow = Wv16 + (ct * 16 + m) * C_DIM + 8 * lh;
#pragma unroll
            for (int c0 = 0; c0 < 4; ++c0)
                acc = wmma(ld_a(xrow + c0 * 32), ld_a(wrow + c0 * 32), acc);
            v8h st;
#pragma unroll
            for (int r = 0; r < 8; ++r) st[r] = (h16)acc[r];
            h16* dst = Vc + ((size_t)(b * C_DIM + ct * 16 + m)) * N_POS +
                       i0 + it * 16 + 8 * lh;
            *(v8h*)dst = st;
        }
    }
}

// ---------------------------------------------------------------------------
// Kernel 2: flash attention. 4 waves/WG, 16 queries/wave, d=128 resident O.
// Scores computed TRANSPOSED (T = Ktile^T * Qtile) so each lane owns exactly
// the 16 P values its PV A-fragment needs: softmax has no LDS round-trip,
// only 1 shfl_xor(16) per reduction. Double-buffered K/V staging.
// ---------------------------------------------------------------------------
__global__ __launch_bounds__(128) void flash_attn_kernel(
    const h16* __restrict__ Qt, const h16* __restrict__ Kt,
    const h16* __restrict__ Vc, h16* __restrict__ Ot)
{
    const int b    = blockIdx.y;
    const int lane = threadIdx.x & 31;
    const int wave = threadIdx.x >> 5;
    const int i0   = blockIdx.x * 64 + wave * 16;
    const int m    = lane & 15;
    const int lh   = lane >> 4;

    __shared__ h16 ks[2][32 * KT_STRIDE];    // [j][c]  position-major
    __shared__ h16 vs[2][C_DIM * VC_STRIDE]; // [c][j]  channel-major

    // resident Q B-fragments: b[e] = Qt[i0+m][32*c0 + 8*lh + e + 8*(e>=8)]
    v16h qb[4];
    {
        const h16* qrow = Qt + ((size_t)(b * N_POS + i0 + m)) * C_DIM + 8 * lh;
#pragma unroll
        for (int c0 = 0; c0 < 4; ++c0) qb[c0] = ld_a(qrow + c0 * 32);
    }

    v8f o[8];
#pragma unroll
    for (int t = 0; t < 8; ++t) o[t] = (v8f){0.f,0.f,0.f,0.f,0.f,0.f,0.f,0.f};
    float mrow = -INFINITY, lrow = 0.0f;     // lane owns row m stats

    const h16* kbase = Kt + (size_t)b * N_POS * C_DIM;
    const h16* vbase = Vc + (size_t)b * C_DIM * N_POS;

    // prologue: stage tile 0 into buffer 0
    {
        const uint4* ksrc = (const uint4*)kbase;
#pragma unroll
        for (int u = 0; u < 4; ++u) {
            int t = threadIdx.x + u * 128;
            int j = t >> 4, q = t & 15;
            *(uint4*)(ks[0] + j * KT_STRIDE + q * 8) = ksrc[t];
            int c = t >> 2, qq = t & 3;
            *(uint4*)(vs[0] + c * VC_STRIDE + qq * 8) =
                *(const uint4*)(vbase + (size_t)c * N_POS + qq * 8);
        }
    }
    __syncthreads();

    int buf = 0;
    for (int j0 = 0; j0 < N_POS; j0 += 32) {
        // ---- issue next tile's global loads early (hidden behind compute) ----
        uint4 kreg[4], vreg[4];
        const bool have_next = (j0 + 32 < N_POS);
        if (have_next) {
            const uint4* ksrc = (const uint4*)(kbase + (size_t)(j0 + 32) * C_DIM);
            const h16*   vsrc = vbase + (j0 + 32);
#pragma unroll
            for (int u = 0; u < 4; ++u) {
                int t = threadIdx.x + u * 128;
                kreg[u] = ksrc[t];
                int c = t >> 2, q = t & 3;
                vreg[u] = *(const uint4*)(vsrc + (size_t)c * N_POS + q * 8);
            }
        }
        const h16* ksb = ks[buf];
        const h16* vsb = vs[buf];

        // ---- T = Ktile^T * Qtile: lane (m,lh) gets S[row m][16 owned cols] ----
        float t01[16];
#pragma unroll
        for (int jt = 0; jt < 2; ++jt) {
            v16h kf[4];
            const h16* kr = ksb + (jt * 16 + m) * KT_STRIDE + 8 * lh;
#pragma unroll
            for (int c0 = 0; c0 < 4; ++c0) kf[c0] = ld_a(kr + c0 * 32);
            v8f acc = (v8f){0.f,0.f,0.f,0.f,0.f,0.f,0.f,0.f};
#pragma unroll
            for (int c0 = 0; c0 < 4; ++c0) acc = wmma(kf[c0], qb[c0], acc);
#pragma unroll
            for (int r = 0; r < 8; ++r) t01[jt * 8 + r] = acc[r];
        }

        // ---- online softmax, lane-owned row ----
        float rmax = t01[0];
#pragma unroll
        for (int k = 1; k < 16; ++k) rmax = fmaxf(rmax, t01[k]);
        rmax = fmaxf(rmax, __shfl_xor(rmax, 16, 32));   // combine halves
        float mn  = fmaxf(mrow, rmax);
        bool  upd = mn > mrow;
        float alpha = __expf(mrow - mn);
        mrow = mn;
        float rs = 0.0f;
        v16h pa;
#pragma unroll
        for (int k = 0; k < 16; ++k) {
            float pv = __expf(t01[k] - mn);
            rs += pv;
            pa[k] = (h16)pv;    // A-layout element order == t01 order by design
        }
        rs += __shfl_xor(rs, 16, 32);
        lrow = lrow * alpha + rs;

        // ---- rescale O only when some row max moved (uniform branch) ----
        if (__ballot(upd)) {
#pragma unroll
            for (int r = 0; r < 8; ++r) {
                float ar = __shfl(alpha, r + 8 * lh, 32);  // row r+8lh owner
#pragma unroll
                for (int ct = 0; ct < 8; ++ct) o[ct][r] *= ar;
            }
        }

        // ---- O += P * V^T ----
#pragma unroll
        for (int g = 0; g < 2; ++g) {
            v16h vf[4];
#pragma unroll
            for (int q = 0; q < 4; ++q)
                vf[q] = ld_a(vsb + ((g * 4 + q) * 16 + m) * VC_STRIDE + 8 * lh);
#pragma unroll
            for (int q = 0; q < 4; ++q)
                o[g * 4 + q] = wmma(pa, vf[q], o[g * 4 + q]);
        }

        // ---- commit staged tile to the other buffer ----
        if (have_next) {
#pragma unroll
            for (int u = 0; u < 4; ++u) {
                int t = threadIdx.x + u * 128;
                int j = t >> 4, q = t & 15;
                *(uint4*)(ks[buf ^ 1] + j * KT_STRIDE + q * 8) = kreg[u];
                int c = t >> 2, qq = t & 3;
                *(uint4*)(vs[buf ^ 1] + c * VC_STRIDE + qq * 8) = vreg[u];
            }
        }
        __syncthreads();
        buf ^= 1;
    }

    // ---- normalize and store Ot[b][i][c] ----
    float linv[8];
#pragma unroll
    for (int r = 0; r < 8; ++r)
        linv[r] = 1.0f / __shfl(lrow, r + 8 * lh, 32);
    h16* obase = Ot + ((size_t)b * N_POS + i0) * C_DIM;
#pragma unroll
    for (int ct = 0; ct < 8; ++ct)
#pragma unroll
        for (int r = 0; r < 8; ++r)
            obase[(size_t)(r + 8 * lh) * C_DIM + ct * 16 + m] =
                (h16)(o[ct][r] * linv[r]);
}

// ---------------------------------------------------------------------------
// Kernel 3: out = Wo @ O^T + bo + x (fp32 result), WMMA with M=cout, N=pos.
// B-fragments read directly from position-major Ot (contiguous); no LDS.
// ---------------------------------------------------------------------------
__global__ __launch_bounds__(256) void out_proj_kernel(
    const h16* __restrict__ Ot, const h16* __restrict__ Wo16,
    const float* __restrict__ bo, const float* __restrict__ x,
    float* __restrict__ out)
{
    const int b    = blockIdx.y;
    const int lane = threadIdx.x & 31;
    const int wave = threadIdx.x >> 5;      // 8 waves -> co tiles
    const int co0  = wave * 16;
    const int m    = lane & 15;
    const int lh   = lane >> 4;

    v16h af[4];
    {
        const h16* wrow = Wo16 + (co0 + m) * C_DIM + 8 * lh;
#pragma unroll
        for (int c0 = 0; c0 < 4; ++c0) af[c0] = ld_a(wrow + c0 * 32);
    }
    float bor[8];
#pragma unroll
    for (int r = 0; r < 8; ++r) bor[r] = bo[co0 + 8 * lh + r];

#pragma unroll
    for (int it = 0; it < 2; ++it) {
        const int i0 = blockIdx.x * 32 + it * 16;
        const h16* orow = Ot + ((size_t)(b * N_POS + i0 + m)) * C_DIM + 8 * lh;
        v8f acc = (v8f){0.f,0.f,0.f,0.f,0.f,0.f,0.f,0.f};
#pragma unroll
        for (int c0 = 0; c0 < 4; ++c0)
            acc = wmma(af[c0], ld_a(orow + c0 * 32), acc);
#pragma unroll
        for (int r = 0; r < 8; ++r) {
            size_t off = ((size_t)(b * C_DIM + co0 + 8 * lh + r)) * N_POS + i0 + m;
            out[off] = acc[r] + bor[r] + x[off];
        }
    }
}

// ---------------------------------------------------------------------------
extern "C" void kernel_launch(void* const* d_in, const int* in_sizes, int n_in,
                              void* d_out, int out_size, void* d_ws, size_t ws_size,
                              hipStream_t stream) {
    const float* x  = (const float*)d_in[0];
    const float* Wq = (const float*)d_in[1];
    const float* bq = (const float*)d_in[2];
    const float* Wk = (const float*)d_in[3];
    const float* bk = (const float*)d_in[4];
    const float* Wv = (const float*)d_in[5];
    const float* bv = (const float*)d_in[6];
    const float* Wo = (const float*)d_in[7];
    const float* bo = (const float*)d_in[8];

    const int b = in_sizes[0] / (C_DIM * N_POS);   // = 4
    const size_t mat = (size_t)b * N_POS * C_DIM;  // f16 elements per tensor

    h16* Qt   = (h16*)d_ws;
    h16* Kt   = Qt + mat;
    h16* Vc   = Kt + mat;
    h16* Ot   = Vc + mat;
    h16* Wq16 = Ot + mat;
    h16* Wk16 = Wq16 + C_DIM * C_DIM;
    h16* Wv16 = Wk16 + C_DIM * C_DIM;
    h16* Wo16 = Wv16 + C_DIM * C_DIM;

    wconv_kernel<<<dim3(64), dim3(256), 0, stream>>>(Wq, Wk, Wv, Wo,
                                                     Wq16, Wk16, Wv16, Wo16);

    qkv_proj_kernel<<<dim3(N_POS / 64, b), dim3(256), 0, stream>>>(
        x, Wq16, bq, Wk16, bk, Wv16, bv, Qt, Kt, Vc);

    flash_attn_kernel<<<dim3(N_POS / 64, b), dim3(128), 0, stream>>>(
        Qt, Kt, Vc, Ot);

    out_proj_kernel<<<dim3(N_POS / 32, b), dim3(256), 0, stream>>>(
        Ot, Wo16, bo, x, (float*)d_out);
}
